// Net_82197084111227
// MI455X (gfx1250) — compile-verified, matrix operand order checked by reading, not run on hardware
//
#include <hip/hip_runtime.h>
#include <hip/hip_bf16.h>

// ---------------- problem constants ----------------
constexpr int Bb     = 8;
constexpr int NUM_PM = 300;
constexpr int NUM_VM = 1000;
constexpr int D      = 256;
constexpr int H      = 8;
constexpr int DFF    = 1024;
constexpr int NL     = 3;
constexpr int L      = NUM_PM + NUM_VM + 2;   // 1302
constexpr int Lp     = 1312;                  // L padded to multiple of 32
constexpr int ML     = Bb * L;                // 10416 (multiple of 16)
constexpr int QT     = (L + 15) / 16;         // 82 query tiles

typedef __attribute__((ext_vector_type(16))) __bf16        v16bf;
typedef __attribute__((ext_vector_type(8)))  float         v8f;
typedef __attribute__((ext_vector_type(8)))  unsigned int  v8u;

struct U16x16 { unsigned short u[16]; };

__device__ __forceinline__ unsigned short f2bf(float f) {
    union { float f; unsigned u; } c; c.f = f;
    unsigned r = c.u + 0x7FFFu + ((c.u >> 16) & 1u);   // RNE
    return (unsigned short)(r >> 16);
}

// Load a 16-element bf16 WMMA fragment (A pattern / transposed-B pattern).
// base must already include the per-lane row offset and +8 K-shift for the
// upper half-wave.  Element i covers K = (i&7) + (i>>3)*16  (dwords 0..3,8..11),
// which the compiler fuses into two global_load_b128 at byte offsets 0 and 32.
__device__ __forceinline__ v16bf load_frag16(const unsigned short* base) {
    const unsigned int* p = (const unsigned int*)base;
    v8u u;
    u[0] = p[0];  u[1] = p[1];  u[2] = p[2];  u[3] = p[3];
    u[4] = p[8];  u[5] = p[9];  u[6] = p[10]; u[7] = p[11];
    return __builtin_bit_cast(v16bf, u);
}

// ---------------- weight transpose + bf16 convert ----------------
// W is (K,N) row-major fp32 -> Wt is (N,K) row-major bf16
__global__ void conv_t_kernel(const float* __restrict__ W,
                              unsigned short* __restrict__ Wt, int K, int N) {
    int idx = blockIdx.x * blockDim.x + threadIdx.x;
    if (idx >= K * N) return;
    int k = idx % K, n = idx / K;
    Wt[(size_t)n * K + k] = f2bf(W[(size_t)k * N + n]);
}

// ---------------- embedding / token assembly ----------------
__global__ void embed_kernel(const float* __restrict__ vm,      // (B,NUM_VM,7)
                             const float* __restrict__ ns,      // (B,1,1)
                             const float* __restrict__ pm,      // (B,NUM_PM,8)
                             const float* __restrict__ W_pm, const float* __restrict__ b_pm,
                             const float* __restrict__ W_vm, const float* __restrict__ b_vm,
                             const float* __restrict__ critic,  // (1,1,D)
                             float* __restrict__ x) {
    int idx = blockIdx.x * blockDim.x + threadIdx.x;
    if (idx >= ML * D) return;
    int d   = idx % D;
    int row = idx / D;
    int pos = row % L;
    int b   = row / L;
    float v;
    if (pos == 0) {
        v = ns[b];
    } else if (pos <= NUM_PM) {
        int i = pos - 1;
        const float* pr = pm + ((size_t)b * NUM_PM + i) * 8;
        float s = b_pm[d];
        #pragma unroll
        for (int j = 0; j < 8; j++) s += pr[j] * W_pm[j * D + d];
        v = s;
    } else if (pos < L - 1) {
        int i = pos - 1 - NUM_PM;
        const float* vr = vm + ((size_t)b * NUM_VM + i) * 7;
        float s = b_vm[d];
        #pragma unroll
        for (int j = 0; j < 6; j++) s += vr[j] * W_vm[j * D + d];
        v = s;
    } else {
        v = critic[d];
    }
    x[(size_t)row * D + d] = v;
}

// relv: per-position group id; equal => attend.  Sentinels: -1 first token,
// -2 critic token, -3 padding (matches nothing real).
__global__ void relv_kernel(const float* __restrict__ vm, float* __restrict__ relv) {
    int idx = blockIdx.x * blockDim.x + threadIdx.x;
    if (idx >= Bb * Lp) return;
    int pos = idx % Lp, b = idx / Lp;
    float v;
    if (pos == 0)            v = -1.f;
    else if (pos == L - 1)   v = -2.f;
    else if (pos >= L)       v = -3.f;
    else {
        int j = pos - 1;
        v = (j < NUM_PM) ? (float)j
                         : vm[((size_t)b * NUM_VM + (j - NUM_PM)) * 7 + 6];
    }
    relv[(size_t)b * Lp + pos] = v;
}

// ---------------- LayerNorm -> bf16 ----------------
__global__ void ln_kernel(const float* __restrict__ x,
                          const float* __restrict__ g, const float* __restrict__ bta,
                          unsigned short* __restrict__ h, int M) {
    int wid  = (blockIdx.x * blockDim.x + threadIdx.x) >> 5;
    int lane = threadIdx.x & 31;
    if (wid >= M) return;
    const float* row = x + (size_t)wid * D;
    float v[8], s = 0.f, ss = 0.f;
    #pragma unroll
    for (int i = 0; i < 8; i++) {
        v[i] = row[lane + 32 * i];
        s += v[i]; ss += v[i] * v[i];
    }
    #pragma unroll
    for (int m = 16; m >= 1; m >>= 1) { s += __shfl_xor(s, m, 32); ss += __shfl_xor(ss, m, 32); }
    float mean = s * (1.f / D);
    float var  = ss * (1.f / D) - mean * mean;
    float inv  = rsqrtf(var + 1e-5f);
    unsigned short* hr = h + (size_t)wid * D;
    #pragma unroll
    for (int i = 0; i < 8; i++) {
        int d = lane + 32 * i;
        hr[d] = f2bf((v[i] - mean) * inv * g[d] + bta[d]);
    }
}

// ---------------- register-blocked bf16 WMMA GEMM ----------------
// out = A(MxK) @ Wt(NxK)^T + bias.  Each wave computes a 16x64 strip: one A
// fragment is reused against 4 contiguous B fragments (4 accumulators), so a
// 32-K step does 5 fragment loads for 4 WMMAs (~13 flops per L2 byte vs 8 for
// an unblocked 16x16 tile).  N must be a multiple of 64 (256/1024 both are).
// epi 0: write bf16     epi 1: gelu(tanh) -> bf16     epi 2: xres += result (fp32)
__global__ void gemm_wmma_kernel(const unsigned short* __restrict__ A,
                                 const unsigned short* __restrict__ Wt,
                                 const float* __restrict__ bias,
                                 float* __restrict__ xres,
                                 unsigned short* __restrict__ outH,
                                 int M, int N, int Kd, int epi) {
    int wid  = (blockIdx.x * blockDim.x + threadIdx.x) >> 5;
    int lane = threadIdx.x & 31;
    int mt = M >> 4, ng = N >> 6;
    if (wid >= mt * ng) return;
    int tm = wid % mt, tg = wid / mt;
    int half = lane >> 4, ln = lane & 15;

    const unsigned short* arow = A  + (size_t)(tm * 16 + ln) * Kd + half * 8;
    const unsigned short* brow = Wt + (size_t)(tg * 64 + ln) * Kd + half * 8;

    v8f acc[4] = {v8f{}, v8f{}, v8f{}, v8f{}};
    for (int kb = 0; kb < Kd; kb += 32) {
        v16bf af = load_frag16(arow + kb);
        #pragma unroll
        for (int j = 0; j < 4; j++) {
            v16bf bf = load_frag16(brow + (size_t)j * 16 * Kd + kb);
            acc[j] = __builtin_amdgcn_wmma_f32_16x16x32_bf16(false, af, false, bf,
                                                             (short)0, acc[j], false, false);
        }
    }
    #pragma unroll
    for (int j = 0; j < 4; j++) {
        int col = tg * 64 + j * 16 + ln;
        float bi = bias ? bias[col] : 0.f;
        #pragma unroll
        for (int r = 0; r < 8; r++) {
            int row = tm * 16 + r + half * 8;
            float v = acc[j][r] + bi;
            size_t o = (size_t)row * N + col;
            if (epi == 2) {
                xres[o] += v;
            } else {
                if (epi == 1) { // tanh gelu
                    float c = 0.7978845608028654f * (v + 0.044715f * v * v * v);
                    v = 0.5f * v * (1.f + tanhf(c));
                }
                outH[o] = f2bf(v);
            }
        }
    }
}

// ---------------- flash attention (one wave per 16-query tile) ----------------
__global__ __launch_bounds__(128)
void attn_kernel(const unsigned short* __restrict__ Q,
                 const unsigned short* __restrict__ Kb,
                 const unsigned short* __restrict__ Vb,
                 const float* __restrict__ relv,
                 unsigned short* __restrict__ O) {
    __shared__ unsigned short ldsP[4][16 * 32];   // per-wave P staging tile
    int wib  = threadIdx.x >> 5;
    int wid  = blockIdx.x * 4 + wib;
    int lane = threadIdx.x & 31;
    int half = lane >> 4, ln = lane & 15;

    int qt = wid % QT; int t = wid / QT;
    int hd = t % H;    int b = t / H;
    const float scale = 0.17677669529663687f;     // 1/sqrt(32)

    // Q fragment: A operand, K-dim = head dim (32) -> single WMMA per score tile
    int qrowA = qt * 16 + ln; if (qrowA > L - 1) qrowA = L - 1;
    v16bf qf = load_frag16(Q + ((size_t)(b * L + qrowA) * D + hd * 32 + half * 8));

    float relq[8];
    #pragma unroll
    for (int r = 0; r < 8; r++) relq[r] = relv[(size_t)b * Lp + qt * 16 + r + half * 8];

    float mrun[8], lsum[8];
    v8f o0 = {}, o1 = {};
    #pragma unroll
    for (int r = 0; r < 8; r++) { mrun[r] = -3.4e38f; lsum[r] = 0.f; }

    unsigned short* P = ldsP[wib];

    for (int kc = 0; kc < Lp; kc += 32) {
        float sv[2][8];
        #pragma unroll
        for (int j = 0; j < 2; j++) {
            int key  = kc + j * 16 + ln;
            int krow = key > L - 1 ? L - 1 : key;
            v16bf kf = load_frag16(Kb + ((size_t)(b * L + krow) * D + hd * 32 + half * 8));
            v8f s = {};
            s = __builtin_amdgcn_wmma_f32_16x16x32_bf16(false, qf, false, kf,
                                                        (short)0, s, false, false);
            float relk = relv[(size_t)b * Lp + key];
            #pragma unroll
            for (int r = 0; r < 8; r++) {
                float val = s[r] * scale;
                if (relq[r] != relk) val = -1e9f;   // mask, as in reference
                sv[j][r] = val;
            }
        }
        // online softmax over this 32-key chunk (row lives in 16 lanes of one half)
        #pragma unroll
        for (int r = 0; r < 8; r++) {
            float mx = fmaxf(sv[0][r], sv[1][r]);
            #pragma unroll
            for (int m = 8; m >= 1; m >>= 1) mx = fmaxf(mx, __shfl_xor(mx, m, 32));
            mx = fmaxf(mx, mrun[r]);
            float corr = __expf(mrun[r] - mx);
            float p0 = __expf(sv[0][r] - mx);
            float p1 = __expf(sv[1][r] - mx);
            float ps = p0 + p1;
            #pragma unroll
            for (int m = 8; m >= 1; m >>= 1) ps += __shfl_xor(ps, m, 32);
            lsum[r] = lsum[r] * corr + ps;
            mrun[r] = mx;
            o0[r] *= corr; o1[r] *= corr;
            int mrow = r + half * 8;                // D-layout -> LDS (row-major 16x32)
            P[mrow * 32 +      ln] = f2bf(p0);
            P[mrow * 32 + 16 + ln] = f2bf(p1);
        }
        asm volatile("s_wait_dscnt 0" ::: "memory");        // LDS transpose hand-off
        v16bf pf = load_frag16(P + ln * 32 + half * 8);      // P as A operand (K=keys)

        // V as B operand: reduction dim = keys (strided rows -> scalar gathers)
        #pragma unroll
        for (int seg = 0; seg < 2; seg++) {
            U16x16 uv;
            #pragma unroll
            for (int e = 0; e < 16; e++) {
                int k    = (e & 7) + (e >> 3) * 16 + half * 8;
                int krow = kc + k; if (krow > L - 1) krow = L - 1;
                uv.u[e] = Vb[(size_t)(b * L + krow) * D + hd * 32 + seg * 16 + ln];
            }
            v16bf vf = __builtin_bit_cast(v16bf, uv);
            if (seg == 0)
                o0 = __builtin_amdgcn_wmma_f32_16x16x32_bf16(false, pf, false, vf,
                                                             (short)0, o0, false, false);
            else
                o1 = __builtin_amdgcn_wmma_f32_16x16x32_bf16(false, pf, false, vf,
                                                             (short)0, o1, false, false);
        }
    }
    #pragma unroll
    for (int r = 0; r < 8; r++) {
        int qi = qt * 16 + r + half * 8;
        if (qi < L) {
            float invl = 1.f / lsum[r];
            size_t o = (size_t)(b * L + qi) * D + hd * 32;
            O[o +      ln] = f2bf(o0[r] * invl);
            O[o + 16 + ln] = f2bf(o1[r] * invl);
        }
    }
}

// ---------------- final scoring head ----------------
__global__ void out_kernel(const float* __restrict__ x, const float* __restrict__ W_out,
                           const float* __restrict__ b_out, float* __restrict__ out) {
    int wid  = (blockIdx.x * blockDim.x + threadIdx.x) >> 5;
    int lane = threadIdx.x & 31;
    if (wid >= Bb * NUM_VM) return;
    int b = wid / NUM_VM, i = wid % NUM_VM;
    const float* row = x + (size_t)(b * L + NUM_PM + 1 + i) * D;
    float s = 0.f;
    for (int d = lane; d < D; d += 32) s += row[d] * W_out[d];
    #pragma unroll
    for (int m = 16; m >= 1; m >>= 1) s += __shfl_xor(s, m, 32);
    if (lane == 0) out[b * NUM_VM + i] = s + b_out[0];
}

// ---------------- host orchestration ----------------
static inline int blocks_for(long threads, int bs) { return (int)((threads + bs - 1) / bs); }

extern "C" void kernel_launch(void* const* d_in, const int* in_sizes, int n_in,
                              void* d_out, int out_size, void* d_ws, size_t ws_size,
                              hipStream_t stream) {
    const float* vm     = (const float*)d_in[0];
    const float* ns     = (const float*)d_in[1];
    const float* pm     = (const float*)d_in[2];
    const float* W_pm   = (const float*)d_in[3];
    const float* b_pm   = (const float*)d_in[4];
    const float* W_vm   = (const float*)d_in[5];
    const float* b_vm   = (const float*)d_in[6];
    const float* critic = (const float*)d_in[7];
    const float* ln1_g  = (const float*)d_in[8];
    const float* ln1_b  = (const float*)d_in[9];
    const float* Wq     = (const float*)d_in[10];
    const float* bq     = (const float*)d_in[11];
    const float* Wk     = (const float*)d_in[12];
    const float* bk     = (const float*)d_in[13];
    const float* Wv     = (const float*)d_in[14];
    const float* bv     = (const float*)d_in[15];
    const float* Wo     = (const float*)d_in[16];
    const float* bo     = (const float*)d_in[17];
    const float* ln2_g  = (const float*)d_in[18];
    const float* ln2_b  = (const float*)d_in[19];
    const float* W1     = (const float*)d_in[20];
    const float* b1     = (const float*)d_in[21];
    const float* W2     = (const float*)d_in[22];
    const float* b2     = (const float*)d_in[23];
    const float* W_out  = (const float*)d_in[24];
    const float* b_out  = (const float*)d_in[25];
    float* outp = (float*)d_out;

    // workspace layout
    char* w = (char*)d_ws;
    size_t off = 0;
    auto carve = [&](size_t bytes) { void* p = w + off; off += (bytes + 255) & ~(size_t)255; return p; };
    float*          x    = (float*)carve((size_t)ML * D * 4);
    unsigned short* h    = (unsigned short*)carve((size_t)ML * D * 2);
    // Q/K/V/O region aliases the FFN gelu buffer (4*ML*D == ML*DFF)
    unsigned short* qkvo = (unsigned short*)carve((size_t)ML * DFF * 2);
    unsigned short* Qb = qkvo;
    unsigned short* Kb = qkvo + (size_t)ML * D;
    unsigned short* Vb = qkvo + (size_t)2 * ML * D;
    unsigned short* Ob = qkvo + (size_t)3 * ML * D;
    unsigned short* g  = qkvo;                      // alias (disjoint lifetime)
    float*          relv = (float*)carve((size_t)Bb * Lp * 4);
    unsigned short* wts  = (unsigned short*)carve((size_t)NL * (4 * D * D + 2 * D * DFF) * 2);
    const size_t perLayer = 4 * (size_t)D * D + 2 * (size_t)D * DFF;

    // weight conversion + transpose (fp32 -> bf16, (K,N) -> (N,K))
    for (int l = 0; l < NL; l++) {
        unsigned short* base = wts + (size_t)l * perLayer;
        conv_t_kernel<<<blocks_for((long)D * D, 256), 256, 0, stream>>>(Wq + (size_t)l * D * D, base,             D, D);
        conv_t_kernel<<<blocks_for((long)D * D, 256), 256, 0, stream>>>(Wk + (size_t)l * D * D, base + D * D,     D, D);
        conv_t_kernel<<<blocks_for((long)D * D, 256), 256, 0, stream>>>(Wv + (size_t)l * D * D, base + 2 * D * D, D, D);
        conv_t_kernel<<<blocks_for((long)D * D, 256), 256, 0, stream>>>(Wo + (size_t)l * D * D, base + 3 * D * D, D, D);
        conv_t_kernel<<<blocks_for((long)D * DFF, 256), 256, 0, stream>>>(W1 + (size_t)l * D * DFF, base + 4 * D * D,           D,   DFF);
        conv_t_kernel<<<blocks_for((long)D * DFF, 256), 256, 0, stream>>>(W2 + (size_t)l * DFF * D, base + 4 * D * D + D * DFF, DFF, D);
    }

    // token assembly + relation table
    embed_kernel<<<blocks_for((long)ML * D, 256), 256, 0, stream>>>(vm, ns, pm, W_pm, b_pm, W_vm, b_vm, critic, x);
    relv_kernel<<<blocks_for((long)Bb * Lp, 256), 256, 0, stream>>>(vm, relv);

    const int gemmTilesD   = (ML / 16) * (D / 64);     // 16x64 strips
    const int gemmTilesDFF = (ML / 16) * (DFF / 64);

    for (int l = 0; l < NL; l++) {
        unsigned short* wq_t = wts + (size_t)l * perLayer;
        unsigned short* wk_t = wq_t + D * D;
        unsigned short* wv_t = wq_t + 2 * D * D;
        unsigned short* wo_t = wq_t + 3 * D * D;
        unsigned short* w1_t = wq_t + 4 * D * D;
        unsigned short* w2_t = wq_t + 4 * D * D + D * DFF;

        // LN1 -> h (bf16)
        ln_kernel<<<blocks_for((long)ML * 32, 256), 256, 0, stream>>>(x, ln1_g + l * D, ln1_b + l * D, h, ML);
        // Q, K, V projections
        gemm_wmma_kernel<<<blocks_for((long)gemmTilesD * 32, 256), 256, 0, stream>>>(h, wq_t, bq + l * D, nullptr, Qb, ML, D, D, 0);
        gemm_wmma_kernel<<<blocks_for((long)gemmTilesD * 32, 256), 256, 0, stream>>>(h, wk_t, bk + l * D, nullptr, Kb, ML, D, D, 0);
        gemm_wmma_kernel<<<blocks_for((long)gemmTilesD * 32, 256), 256, 0, stream>>>(h, wv_t, bv + l * D, nullptr, Vb, ML, D, D, 0);
        // flash attention -> Ob
        attn_kernel<<<(Bb * H * QT) / 4, 128, 0, stream>>>(Qb, Kb, Vb, relv, Ob);
        // x += Ob @ Wo + bo
        gemm_wmma_kernel<<<blocks_for((long)gemmTilesD * 32, 256), 256, 0, stream>>>(Ob, wo_t, bo + l * D, x, nullptr, ML, D, D, 2);
        // LN2 -> h
        ln_kernel<<<blocks_for((long)ML * 32, 256), 256, 0, stream>>>(x, ln2_g + l * D, ln2_b + l * D, h, ML);
        // g = gelu(h @ W1 + b1)
        gemm_wmma_kernel<<<blocks_for((long)gemmTilesDFF * 32, 256), 256, 0, stream>>>(h, w1_t, b1 + l * DFF, nullptr, g, ML, DFF, D, 1);
        // x += g @ W2 + b2
        gemm_wmma_kernel<<<blocks_for((long)gemmTilesD * 32, 256), 256, 0, stream>>>(g, w2_t, b2 + l * D, x, nullptr, ML, D, DFF, 2);
    }

    out_kernel<<<blocks_for((long)Bb * NUM_VM * 32, 256), 256, 0, stream>>>(x, W_out, b_out, outp);
    (void)in_sizes; (void)n_in; (void)out_size; (void)ws_size;
}
